// LayerNormMLP_41455024341174
// MI455X (gfx1250) — compile-verified, matrix operand order checked by reading, not run on hardware
//
#include <hip/hip_runtime.h>
#include <stdint.h>
#include <math.h>

// ---- vector types (trivial ext-vectors so unions are legal) ----
typedef __attribute__((ext_vector_type(16))) __bf16 v16bf;
typedef __attribute__((ext_vector_type(8)))  float  v8f;
typedef __attribute__((ext_vector_type(4)))  float  f32x4;
typedef __attribute__((ext_vector_type(4)))  unsigned int   u32x4;
typedef __attribute__((ext_vector_type(4)))  unsigned short u16x4;

__device__ __forceinline__ uint16_t f32_to_bf16(float f) {
  union { float f; uint32_t u; } c; c.f = f;
  uint32_t u = c.u;
  return (uint16_t)((u + 0x7FFFu + ((u >> 16) & 1u)) >> 16);
}

// gfx1250 async global->LDS copy, 16B per lane, tracked by ASYNCcnt.
__device__ __forceinline__ void async_copy_b128(uint32_t lds_off, const void* gptr) {
  asm volatile("global_load_async_to_lds_b128 %0, %1, off"
               :: "v"(lds_off), "v"((uint64_t)(uintptr_t)gptr) : "memory");
}
__device__ __forceinline__ uint32_t lds_off_u32(const void* p) {
  return (uint32_t)(uintptr_t)p;   // LDS aperture: addr[31:0] == LDS byte offset
}

// ---------------- LayerNorm (fp32 in -> bf16 out) ----------------
__global__ __launch_bounds__(256)
void ln_bf16_kernel(const float* __restrict__ x, const float* __restrict__ scale,
                    const float* __restrict__ bias, uint16_t* __restrict__ yb, int H) {
  const int row = blockIdx.x;
  const int t   = threadIdx.x;
  const float* xr = x + (size_t)row * H;
  uint16_t*    yr = yb + (size_t)row * H;

  float sum = 0.f, sumsq = 0.f;
  const int n4 = H >> 2;
  for (int i = t; i < n4; i += 256) {
    f32x4 v = ((const f32x4*)xr)[i];
    #pragma unroll
    for (int j = 0; j < 4; ++j) { sum += v[j]; sumsq += v[j] * v[j]; }
  }
  __shared__ float rs[256], rq[256];
  rs[t] = sum; rq[t] = sumsq;
  __syncthreads();
  for (int s = 128; s > 0; s >>= 1) {
    if (t < s) { rs[t] += rs[t + s]; rq[t] += rq[t + s]; }
    __syncthreads();
  }
  const float invH = 1.0f / (float)H;
  const float mu   = rs[0] * invH;
  const float var  = rq[0] * invH - mu * mu;
  const float rstd = rsqrtf(var + 1e-6f);

  for (int i = t; i < n4; i += 256) {
    f32x4 v  = ((const f32x4*)xr)[i];
    f32x4 sc = ((const f32x4*)scale)[i];
    f32x4 bv = ((const f32x4*)bias)[i];
    u16x4 o;
    #pragma unroll
    for (int j = 0; j < 4; ++j)
      o[j] = f32_to_bf16((v[j] - mu) * rstd * sc[j] + bv[j]);
    ((u16x4*)yr)[i] = o;
  }
}

// -------- fp32 [R][C] -> bf16 transposed [C][R] (one-time weight prep) --------
__global__ __launch_bounds__(256)
void cvt_transpose_bf16_kernel(const float* __restrict__ in, uint16_t* __restrict__ out,
                               int R, int C) {
  __shared__ uint16_t tile[32][33];
  const int t  = threadIdx.x;
  const int tx = t & 31, ty = t >> 5;                // ty = 0..7
  const int r0 = blockIdx.y * 32, c0 = blockIdx.x * 32;
  #pragma unroll
  for (int j = 0; j < 4; ++j) {
    int r = r0 + ty + 8 * j;
    tile[ty + 8 * j][tx] = f32_to_bf16(in[(size_t)r * C + c0 + tx]);
  }
  __syncthreads();
  #pragma unroll
  for (int j = 0; j < 4; ++j) {
    int c = c0 + ty + 8 * j;
    out[(size_t)c * R + r0 + tx] = tile[tx][ty + 8 * j];
  }
}

// ---------------- WMMA helpers ----------------
// 16-bit fragment (16 rows x K=32): lane l in [0,15] holds row l,
// K = half*8..half*8+7 (chunk0) and 16+half*8..+7 (chunk1), half = lane>>4.
union FragU { u32x4 q[2]; v16bf v; };

__device__ __forceinline__ v16bf load_frag(const uint16_t* p) {
  FragU f;
  f.q[0] = *(const u32x4*)(p);
  f.q[1] = *(const u32x4*)(p + 16);
  return f.v;
}

#define BM  128
#define BN  64
#define BK  32
#define LDA 40   // 32 + 8 pad; 80B row = 5 x 16B chunks
#define LDB 40

// ------- GEMM1 + GeGLU: z = gelu(y@W1[:,0,:]) * (y@W1[:,1,:]), W1 pre-transposed [2I][H] -------
__global__ __launch_bounds__(256)
void gemm1_geglu_kernel(const uint16_t* __restrict__ yb, const uint16_t* __restrict__ w1t,
                        uint16_t* __restrict__ zb, int M, int H, int I) {
  __shared__ uint16_t sA [2][BM * LDA];
  __shared__ uint16_t sB0[2][BN * LDB];
  __shared__ uint16_t sB1[2][BN * LDB];

  const int t    = threadIdx.x;
  const int lane = t & 31;
  const int wave = t >> 5;
  const int m0 = blockIdx.y * BM;
  const int n0 = blockIdx.x * BN;
  const int waveM = (wave >> 1) * 32;  // 0,32,64,96
  const int waveN = (wave & 1)  * 32;  // 0,32
  const int r    = lane & 15;
  const int half = lane >> 4;

  const v8f vzero = {0.f,0.f,0.f,0.f,0.f,0.f,0.f,0.f};
  v8f acc0[2][2], acc1[2][2];
  #pragma unroll
  for (int i = 0; i < 2; ++i)
    #pragma unroll
    for (int j = 0; j < 2; ++j) { acc0[i][j] = vzero; acc1[i][j] = vzero; }

  const size_t wOff1 = (size_t)I * H;     // branch-1 offset in transposed W1
  const int kBlocks  = H / BK;

  // per-thread copy coordinates
  const int aR0 = t >> 2,        aC0 = (t & 3) * 8;          // + second chunk at idx+256
  const int aR1 = (t + 256) >> 2, aC1 = ((t + 256) & 3) * 8;
  const int bR  = t >> 2,        bC  = (t & 3) * 8;          // 64 rows x 4 chunks

  auto issue = [&](int buf, int kk) {
    async_copy_b128(lds_off_u32(&sA[buf][aR0 * LDA + aC0]),
                    yb + (size_t)(m0 + aR0) * H + kk + aC0);
    async_copy_b128(lds_off_u32(&sA[buf][aR1 * LDA + aC1]),
                    yb + (size_t)(m0 + aR1) * H + kk + aC1);
    async_copy_b128(lds_off_u32(&sB0[buf][bR * LDB + bC]),
                    w1t + (size_t)(n0 + bR) * H + kk + bC);
    async_copy_b128(lds_off_u32(&sB1[buf][bR * LDB + bC]),
                    w1t + wOff1 + (size_t)(n0 + bR) * H + kk + bC);
  };

  issue(0, 0);  // prologue fill of buffer 0

  for (int kb = 0; kb < kBlocks; ++kb) {
    const int buf   = kb & 1;
    const int kNext = (kb + 1 < kBlocks) ? (kb + 1) * BK : 0;  // clamp keeps count uniform
    __syncthreads();                         // nobody still reading buf^1
    issue(buf ^ 1, kNext);                   // prefetch next tile
    asm volatile("s_wait_asynccnt 0x4" ::: "memory");  // own fills of `buf` landed
    __syncthreads();                         // all waves' fills of `buf` landed

    v16bf aF[2], b0F[2], b1F[2];
    #pragma unroll
    for (int mt = 0; mt < 2; ++mt)
      aF[mt] = load_frag(&sA[buf][(waveM + mt * 16 + r) * LDA + half * 8]);
    #pragma unroll
    for (int nt = 0; nt < 2; ++nt) {
      b0F[nt] = load_frag(&sB0[buf][(waveN + nt * 16 + r) * LDB + half * 8]);
      b1F[nt] = load_frag(&sB1[buf][(waveN + nt * 16 + r) * LDB + half * 8]);
    }
    #pragma unroll
    for (int mt = 0; mt < 2; ++mt)
      #pragma unroll
      for (int nt = 0; nt < 2; ++nt) {
        acc0[mt][nt] = __builtin_amdgcn_wmma_f32_16x16x32_bf16(
            false, aF[mt], false, b0F[nt], (short)0, acc0[mt][nt], false, false);
        acc1[mt][nt] = __builtin_amdgcn_wmma_f32_16x16x32_bf16(
            false, aF[mt], false, b1F[nt], (short)0, acc1[mt][nt], false, false);
      }
  }

  // epilogue: z = gelu_tanh(h0) * h1 -> bf16   (branch-free: 0.5x(1+tanh(u)) = x*sigmoid(2u))
  #pragma unroll
  for (int mt = 0; mt < 2; ++mt)
    #pragma unroll
    for (int nt = 0; nt < 2; ++nt)
      #pragma unroll
      for (int i = 0; i < 8; ++i) {
        int row = m0 + waveM + mt * 16 + i + 8 * half;
        int col = n0 + waveN + nt * 16 + r;
        float h0 = acc0[mt][nt][i];
        float h1 = acc1[mt][nt][i];
        float u  = 1.5957691216057308f * (h0 + 0.044715f * h0 * h0 * h0); // 2*sqrt(2/pi)*(...)
        float g  = h0 * __fdividef(1.0f, 1.0f + __expf(-u));
        zb[(size_t)row * I + col] = f32_to_bf16(g * h1);
      }
}

// ------- GEMM2: out = z @ W2, W2 pre-transposed [H][I], f32 out -------
__global__ __launch_bounds__(256)
void gemm2_kernel(const uint16_t* __restrict__ zb, const uint16_t* __restrict__ w2t,
                  float* __restrict__ out, int M, int H, int I) {
  __shared__ uint16_t sA[2][BM * LDA];
  __shared__ uint16_t sB[2][BN * LDB];

  const int t    = threadIdx.x;
  const int lane = t & 31;
  const int wave = t >> 5;
  const int m0 = blockIdx.y * BM;
  const int n0 = blockIdx.x * BN;
  const int waveM = (wave >> 1) * 32;
  const int waveN = (wave & 1)  * 32;
  const int r    = lane & 15;
  const int half = lane >> 4;

  const v8f vzero = {0.f,0.f,0.f,0.f,0.f,0.f,0.f,0.f};
  v8f acc[2][2];
  #pragma unroll
  for (int i = 0; i < 2; ++i)
    #pragma unroll
    for (int j = 0; j < 2; ++j) acc[i][j] = vzero;

  const int kBlocks = I / BK;
  const int aR0 = t >> 2,         aC0 = (t & 3) * 8;
  const int aR1 = (t + 256) >> 2, aC1 = ((t + 256) & 3) * 8;
  const int bR  = t >> 2,         bC  = (t & 3) * 8;

  auto issue = [&](int buf, int kk) {
    async_copy_b128(lds_off_u32(&sA[buf][aR0 * LDA + aC0]),
                    zb + (size_t)(m0 + aR0) * I + kk + aC0);
    async_copy_b128(lds_off_u32(&sA[buf][aR1 * LDA + aC1]),
                    zb + (size_t)(m0 + aR1) * I + kk + aC1);
    async_copy_b128(lds_off_u32(&sB[buf][bR * LDB + bC]),
                    w2t + (size_t)(n0 + bR) * I + kk + bC);
  };

  issue(0, 0);

  for (int kb = 0; kb < kBlocks; ++kb) {
    const int buf   = kb & 1;
    const int kNext = (kb + 1 < kBlocks) ? (kb + 1) * BK : 0;
    __syncthreads();
    issue(buf ^ 1, kNext);
    asm volatile("s_wait_asynccnt 0x3" ::: "memory");
    __syncthreads();

    v16bf aF[2], bF[2];
    #pragma unroll
    for (int mt = 0; mt < 2; ++mt)
      aF[mt] = load_frag(&sA[buf][(waveM + mt * 16 + r) * LDA + half * 8]);
    #pragma unroll
    for (int nt = 0; nt < 2; ++nt)
      bF[nt] = load_frag(&sB[buf][(waveN + nt * 16 + r) * LDB + half * 8]);

    #pragma unroll
    for (int mt = 0; mt < 2; ++mt)
      #pragma unroll
      for (int nt = 0; nt < 2; ++nt)
        acc[mt][nt] = __builtin_amdgcn_wmma_f32_16x16x32_bf16(
            false, aF[mt], false, bF[nt], (short)0, acc[mt][nt], false, false);
  }

  #pragma unroll
  for (int mt = 0; mt < 2; ++mt)
    #pragma unroll
    for (int nt = 0; nt < 2; ++nt)
      #pragma unroll
      for (int i = 0; i < 8; ++i) {
        int row = m0 + waveM + mt * 16 + i + 8 * half;
        int col = n0 + waveN + nt * 16 + r;
        out[(size_t)row * H + col] = acc[mt][nt][i];
      }
}

// ---------------- host launcher ----------------
extern "C" void kernel_launch(void* const* d_in, const int* in_sizes, int n_in,
                              void* d_out, int out_size, void* d_ws, size_t ws_size,
                              hipStream_t stream) {
  const float* x       = (const float*)d_in[0];
  const float* scale   = (const float*)d_in[1];
  const float* ln_bias = (const float*)d_in[2];
  const float* k1      = (const float*)d_in[3];
  const float* k2      = (const float*)d_in[4];
  float* out = (float*)d_out;

  const int H = in_sizes[1];                 // 2048
  const int M = in_sizes[0] / H;             // B*S = 8192
  const int I = in_sizes[3] / (2 * H);       // 8192

  // bf16 workspace layout
  uint16_t* ws  = (uint16_t*)d_ws;
  size_t off = 0;
  uint16_t* yb  = ws + off; off += (size_t)M * H;       // y (LN output)            32 MB
  uint16_t* w1t = ws + off; off += (size_t)2 * I * H;   // W1^T  [2I][H]            64 MB
  uint16_t* w2t = ws + off; off += (size_t)H * I;       // W2^T  [H][I]             32 MB
  uint16_t* zb  = ws + off; off += (size_t)M * I;       // z (GeGLU output)        128 MB

  ln_bf16_kernel<<<M, 256, 0, stream>>>(x, scale, ln_bias, yb, H);
  // k1: [H][2I] -> [2I][H] ; k2: [I][H] -> [H][I]
  cvt_transpose_bf16_kernel<<<dim3((2 * I) / 32, H / 32), 256, 0, stream>>>(k1, w1t, H, 2 * I);
  cvt_transpose_bf16_kernel<<<dim3(H / 32, I / 32), 256, 0, stream>>>(k2, w2t, I, H);

  gemm1_geglu_kernel<<<dim3(I / BN, M / BM), 256, 0, stream>>>(yb, w1t, zb, M, H, I);
  gemm2_kernel<<<dim3(H / BN, M / BM), 256, 0, stream>>>(zb, w2t, out, M, H, I);
}
